// ScaledDotProductAttention_6227702579504
// MI455X (gfx1250) — matrix operand order repaired and checked by reading, not compiled, not run on hardware
//
#include <hip/hip_runtime.h>
#include <math.h>

// ---- fixed problem shape from the reference ----
constexpr int N  = 2048;
constexpr int D  = 64;
constexpr int QT = 128;   // query rows per workgroup (8 waves * 16)
constexpr int KT = 32;    // keys per inner iteration

// LDS strides (in _Float16 elements)
constexpr int QSTR = 80;  // 160B row: 32B aligned
constexpr int KSTR = 80;  // 160B row: 32B aligned
constexpr int VSTR = 40;  // 80B row: 16B aligned (rows indexed by d, cols = position)
constexpr int PSTR = 40;  // 80B row: 16B aligned (cols = position)

typedef __attribute__((ext_vector_type(16))) _Float16 v16h;
typedef __attribute__((ext_vector_type(8)))  _Float16 v8h;
typedef __attribute__((ext_vector_type(2)))  _Float16 h2;
typedef __attribute__((ext_vector_type(8)))  float    v8f;
typedef __attribute__((ext_vector_type(2)))  float    f2;

// pack two f32 -> two f16 in one VALU op (v_cvt_pk_rtz_f16_f32)
static __device__ __forceinline__ h2 pkh(float a, float b) {
    return __builtin_bit_cast(h2, __builtin_amdgcn_cvt_pkrtz(a, b));
}

// ---- single-instruction DPP16 butterflies (stay inside each 16-lane half) ----
// dst = op(dpp(src0), src1) with dst==src0==src1: accumulating butterfly step.
static __device__ __forceinline__ float rowMax16(float v) {
    asm("v_max_num_f32_dpp %0, %0, %0 quad_perm:[1,0,3,2] row_mask:0xf bank_mask:0xf bound_ctrl:1" : "+v"(v));
    asm("v_max_num_f32_dpp %0, %0, %0 quad_perm:[2,3,0,1] row_mask:0xf bank_mask:0xf bound_ctrl:1" : "+v"(v));
    asm("v_max_num_f32_dpp %0, %0, %0 row_half_mirror row_mask:0xf bank_mask:0xf bound_ctrl:1" : "+v"(v));
    asm("v_max_num_f32_dpp %0, %0, %0 row_mirror row_mask:0xf bank_mask:0xf bound_ctrl:1" : "+v"(v));
    return v;
}
static __device__ __forceinline__ float rowSum16(float v) {
    asm("v_add_f32_dpp %0, %0, %0 quad_perm:[1,0,3,2] row_mask:0xf bank_mask:0xf bound_ctrl:1" : "+v"(v));
    asm("v_add_f32_dpp %0, %0, %0 quad_perm:[2,3,0,1] row_mask:0xf bank_mask:0xf bound_ctrl:1" : "+v"(v));
    asm("v_add_f32_dpp %0, %0, %0 row_half_mirror row_mask:0xf bank_mask:0xf bound_ctrl:1" : "+v"(v));
    asm("v_add_f32_dpp %0, %0, %0 row_mirror row_mask:0xf bank_mask:0xf bound_ctrl:1" : "+v"(v));
    return v;
}
static __device__ __forceinline__ float vmax(float a, float b) {
    float r;
    asm("v_max_num_f32_e32 %0, %1, %2" : "=v"(r) : "v"(a), "v"(b));
    return r;
}

__global__ __launch_bounds__(256)
void fa_wmma_f16_kernel(const float* __restrict__ Qg,
                        const float* __restrict__ Kg,
                        const float* __restrict__ Vg,
                        const float* __restrict__ scal,
                        float* __restrict__ Og)
{
    __shared__ _Float16 sQ[QT * QSTR];          // 20.0 KB
    __shared__ _Float16 sK[2][KT * KSTR];       // 10.0 KB   sK[buf][key][d]
    __shared__ _Float16 sVt[2][D * VSTR];       // 10.0 KB   sVt[buf][d][pos]
    __shared__ _Float16 sP[8 * 16 * PSTR];      // 10.0 KB   per-wave P scratch [row][pos]

    // position mapping: pos = 2c   <-> key c       (score tile 0 column c)
    //                   pos = 2c+1 <-> key 16+c    (score tile 1 column c)
    // WMMA reduction index k == pos for both the P A-fragment and the V B-fragment,
    // so fragment reads stay contiguous while P stores become packed b32.

    const int tid      = threadIdx.x;
    const int wave     = tid >> 5;
    const int lane     = tid & 31;
    const int lane16   = lane & 15;
    const int laneHalf = lane >> 4;

    const int nqt = N / QT;
    const int b   = blockIdx.x / nqt;
    const int q0  = (blockIdx.x % nqt) * QT;

    // exp2-domain softmax: fold log2(e) into the Q pre-scale
    const float qscale = (1.0f / scal[0]) * 1.4426950408889634f;
    const size_t batchBase = (size_t)b * N * D;

    // staging maps
    const int key8 = tid >> 5;          // 0..7   (K/Q staging: row group)
    const int d2   = (tid & 31) * 2;    // 0..62  (K/Q staging: d pair)
    const int keyc4 = tid >> 6;         // 0..3   (V staging: key-column group)
    const int stD   = tid & 63;         // 0..63  (V staging: d)

    // ---- stage Q (pre-scaled) into LDS as f16, packed pairs ----
    for (int k = 0; k < QT / 8; ++k) {
        int r = k * 8 + key8;
        f2 q = *(const f2*)&Qg[batchBase + (size_t)(q0 + r) * D + d2];
        *(h2*)&sQ[r * QSTR + d2] = pkh(q.x * qscale, q.y * qscale);
    }
    __syncthreads();

    // ---- resident Q A-fragments: 16x32 f16, two D-chunks ----
    v16h aq0, aq1;
    {
        const _Float16* qrow = &sQ[(wave * 16 + lane16) * QSTR];
        union { v16h v; v8h h[2]; } u0, u1;
        u0.h[0] = *(const v8h*)(qrow +      laneHalf * 8);
        u0.h[1] = *(const v8h*)(qrow + 16 + laneHalf * 8);
        u1.h[0] = *(const v8h*)(qrow + 32 + laneHalf * 8);
        u1.h[1] = *(const v8h*)(qrow + 48 + laneHalf * 8);
        aq0 = u0.v; aq1 = u1.v;
    }

    // ---- flash-attention state ----
    v8f o0 = {}, o1 = {}, o2 = {}, o3 = {};
    float mrow[8], lrow[8];
#pragma unroll
    for (int r = 0; r < 8; ++r) { mrow[r] = -INFINITY; lrow[r] = 0.0f; }

    _Float16* myP = &sP[wave * 16 * PSTR];

    // ---- software pipeline: next K/V block held in registers ----
    f2    kreg[4];         // K: d-pairs for keys key8, key8+8, key8+16, key8+24
    float vlo[4], vhi[4];  // V: rows keyc, keyc+16 at column stD
    {
#pragma unroll
        for (int k = 0; k < 4; ++k) {
            kreg[k] = *(const f2*)&Kg[batchBase + (size_t)(k * 8 + key8) * D + d2];
            int keyc = k * 4 + keyc4;
            vlo[k] = Vg[batchBase + (size_t)keyc * D + stD];
            vhi[k] = Vg[batchBase + (size_t)(keyc + 16) * D + stD];
        }
    }

    constexpr int NB = N / KT;
    for (int i = 0; i < NB; ++i) {
        _Float16* curK = sK[i & 1];
        _Float16* curV = sVt[i & 1];

        __syncthreads();   // buffer (i&1) free (last read: iteration i-2)

        // regs -> LDS (convert + pack here)
#pragma unroll
        for (int k = 0; k < 4; ++k) {
            int key = k * 8 + key8;
            *(h2*)&curK[key * KSTR + d2] = pkh(kreg[k].x, kreg[k].y);
            int keyc = k * 4 + keyc4;
            *(h2*)&curV[stD * VSTR + 2 * keyc] = pkh(vlo[k], vhi[k]);
        }

        // issue next block's global loads; wait lands after this block's compute
        if (i + 1 < NB) {
            const size_t nb0 = batchBase + (size_t)(i + 1) * KT * D;
#pragma unroll
            for (int k = 0; k < 4; ++k) {
                kreg[k] = *(const f2*)&Kg[nb0 + (size_t)(k * 8 + key8) * D + d2];
                int keyc = k * 4 + keyc4;
                vlo[k] = Vg[nb0 + (size_t)keyc * D + stD];
                vhi[k] = Vg[nb0 + (size_t)(keyc + 16) * D + stD];
            }
        }

        __syncthreads();   // staged block visible

        // ---- S = (Q*qscale) K^T : two 16-key column tiles ----
        v8f s0 = {}, s1 = {};
        {
            const _Float16* kc0 = &curK[(lane16     ) * KSTR + laneHalf * 8];
            const _Float16* kc1 = &curK[(16 + lane16) * KSTR + laneHalf * 8];
            union { v16h v; v8h h[2]; } t;
            t.h[0] = *(const v8h*)(kc0);      t.h[1] = *(const v8h*)(kc0 + 16); v16h b00 = t.v;
            t.h[0] = *(const v8h*)(kc0 + 32); t.h[1] = *(const v8h*)(kc0 + 48); v16h b01 = t.v;
            t.h[0] = *(const v8h*)(kc1);      t.h[1] = *(const v8h*)(kc1 + 16); v16h b10 = t.v;
            t.h[0] = *(const v8h*)(kc1 + 32); t.h[1] = *(const v8h*)(kc1 + 48); v16h b11 = t.v;
            s0 = __builtin_amdgcn_wmma_f32_16x16x32_f16(false, aq0, false, b00, (short)0, s0, false, false);
            s0 = __builtin_amdgcn_wmma_f32_16x16x32_f16(false, aq1, false, b01, (short)0, s0, false, false);
            s1 = __builtin_amdgcn_wmma_f32_16x16x32_f16(false, aq0, false, b10, (short)0, s1, false, false);
            s1 = __builtin_amdgcn_wmma_f32_16x16x32_f16(false, aq1, false, b11, (short)0, s1, false, false);
        }

        // ---- online softmax (exp2 domain); C layout: VGPR r -> row, lane16 -> col ----
#pragma unroll
        for (int r = 0; r < 8; ++r) {
            float v0 = s0[r], v1 = s1[r];
            float mx = rowMax16(vmax(v0, v1));           // 5 VALU, no LDS
            float mnew = vmax(mrow[r], mx);
            float corr = exp2f(mrow[r] - mnew);
            float p0 = exp2f(v0 - mnew);
            float p1 = exp2f(v1 - mnew);
            float ps = rowSum16(p0 + p1);                // 4 VALU
            lrow[r] = lrow[r] * corr + ps;
            mrow[r] = mnew;
            o0[r] *= corr; o1[r] *= corr; o2[r] *= corr; o3[r] *= corr;
            int m = laneHalf * 8 + r;
            // (p0,p1) -> positions (2c, 2c+1): single packed b32 store
            *(h2*)&myP[m * PSTR + 2 * lane16] = pkh(p0, p1);
        }

        // ---- P as 16x32 A-fragment over positions (same-wave LDS, in-order) ----
        v16h ap;
        {
            const _Float16* prow = &myP[lane16 * PSTR];
            union { v16h v; v8h h[2]; } u;
            u.h[0] = *(const v8h*)(prow +      laneHalf * 8);
            u.h[1] = *(const v8h*)(prow + 16 + laneHalf * 8);
            ap = u.v;
        }

        // ---- O += P * V : B-fragment rows = positions, cols = d ----
        {
            const _Float16* vb;
            union { v16h v; v8h h[2]; } u;
            vb = &curV[(0 * 16 + lane16) * VSTR + laneHalf * 8];
            u.h[0] = *(const v8h*)(vb); u.h[1] = *(const v8h*)(vb + 16);
            o0 = __builtin_amdgcn_wmma_f32_16x16x32_f16(false, ap, false, u.v, (short)0, o0, false, false);
            vb = &curV[(1 * 16 + lane16) * VSTR + laneHalf * 8];
            u.h[0] = *(const v8h*)(vb); u.h[1] = *(const v8h*)(vb + 16);
            o1 = __builtin_amdgcn_wmma_f32_16x16x32_f16(false, ap, false, u.v, (short)0, o1, false, false);
            vb = &curV[(2 * 16 + lane16) * VSTR + laneHalf * 8];
            u.h[0] = *(const v8h*)(vb); u.h[1] = *(const v8h*)(vb + 16);
            o2 = __builtin_amdgcn_wmma_f32_16x16x32_f16(false, ap, false, u.v, (short)0, o2, false, false);
            vb = &curV[(3 * 16 + lane16) * VSTR + laneHalf * 8];
            u.h[0] = *(const v8h*)(vb); u.h[1] = *(const v8h*)(vb + 16);
            o3 = __builtin_amdgcn_wmma_f32_16x16x32_f16(false, ap, false, u.v, (short)0, o3, false, false);
        }
    }

    // ---- normalize and write O ----
#pragma unroll
    for (int r = 0; r < 8; ++r) {
        float rl = 1.0f / lrow[r];
        int grow = q0 + wave * 16 + laneHalf * 8 + r;
        float* orow = &Og[batchBase + (size_t)grow * D + lane16];
        orow[0]  = o0[r] * rl;
        orow[16] = o1[r] * rl;
        orow[32] = o2[r] * rl;
        orow[48] = o3[r] * rl;
    }
}

extern "C" void kernel_launch(void* const* d_in, const int* in_sizes, int n_in,
                              void* d_out, int out_size, void* d_ws, size_t ws_size,
                              hipStream_t stream) {
    const float* Q    = (const float*)d_in[0];
    const float* K    = (const float*)d_in[1];
    const float* V    = (const float*)d_in[2];
    const float* scal = (const float*)d_in[3];
    float* out = (float*)d_out;

    const int batches = in_sizes[0] / (N * D);   // B = 16
    const int nblocks = batches * (N / QT);      // 256 workgroups
    fa_wmma_f16_kernel<<<nblocks, 256, 0, stream>>>(Q, K, V, scal, out);
}
